// AuxiliaryMultiHeadedAttention_light_23347442221589
// MI455X (gfx1250) — matrix operand order repaired and checked by reading, not compiled, not run on hardware
//
#include <hip/hip_runtime.h>
#include <hip/hip_bf16.h>

#define BB  4
#define LL  2048
#define HH  256
#define NHH 8
#define DHH 32

typedef __attribute__((ext_vector_type(16))) __bf16 v16bf;
typedef __attribute__((ext_vector_type(8)))  __bf16 v8bf;
typedef __attribute__((ext_vector_type(2)))  __bf16 v2bf;
typedef __attribute__((ext_vector_type(8)))  float  v8f;
typedef __attribute__((ext_vector_type(4)))  float  v4f;

#define LOG2E 1.4426950408889634f
#define MASKED_LOG2 (-1.442695e9f)   /* -1e9 * log2(e) */

static __device__ __forceinline__ v8f zero8() {
  v8f z;
#pragma unroll
  for (int i = 0; i < 8; ++i) z[i] = 0.f;
  return z;
}

static __device__ __forceinline__ v8f wmma_bf16(v16bf a, v16bf b, v8f c) {
  // (neg_a, A, neg_b, B, c_mod, C, reuse_a, reuse_b)
  return __builtin_amdgcn_wmma_f32_16x16x32_bf16(false, a, false, b, (short)0, c,
                                                 false, false);
}

// A-fragment (16x32 bf16): lane holds K = {2kb..2kb+15} (contiguous)
static __device__ __forceinline__ v16bf af_from_bf16(const __bf16* base, int kb) {
  v8bf lo = *(const v8bf*)(base + 2 * kb);
  v8bf hi = *(const v8bf*)(base + 2 * kb + 8);
  v16bf r;
#pragma unroll
  for (int i = 0; i < 8; ++i) { r[i] = lo[i]; r[8 + i] = hi[i]; }
  return r;
}

// B-fragment: 16 contiguous K values per lane (bf16)
static __device__ __forceinline__ v16bf bfrag_from_bf16(const __bf16* p) {
  v8bf lo = *(const v8bf*)(p);
  v8bf hi = *(const v8bf*)(p + 8);
  v16bf r;
#pragma unroll
  for (int i = 0; i < 8; ++i) { r[i] = lo[i]; r[8 + i] = hi[i]; }
  return r;
}

// packed bf16 pair from two floats (pattern-matches to v_cvt_pk_bf16_f32)
static __device__ __forceinline__ unsigned pkbf(float a, float b) {
  v2bf t;
  t[0] = (__bf16)a;
  t[1] = (__bf16)b;
  return __builtin_bit_cast(unsigned, t);
}

// Assemble a P^T B-fragment (K = 32 keys, lane = query) from two C-layout
// probability tiles pA (keys +0..15) / pB (keys +16..31). Packed exchange:
// 8 cvt_pk + 8 shfl + 8 select.
static __device__ __forceinline__ v16bf make_pf(const float* pA, const float* pB,
                                                int hi) {
  unsigned c0[4], c1[4];
#pragma unroll
  for (int v = 0; v < 4; ++v) {
    c0[v] = pkbf(pA[2 * v], pA[2 * v + 1]);
    c1[v] = pkbf(pB[2 * v], pB[2 * v + 1]);
  }
  union { v16bf v; unsigned u[8]; } r;
#pragma unroll
  for (int v = 0; v < 4; ++v) {
    const unsigned s0 = __shfl_xor(c0[v], 16, 32);
    const unsigned s1 = __shfl_xor(c1[v], 16, 32);
    r.u[v]     = hi ? s1 : c0[v];   // K = koff + 2v, 2v+1
    r.u[4 + v] = hi ? c1[v] : s0;   // K = koff + 8 + 2v, ...
  }
  return r.v;
}

// ---------------------------------------------------------------------------
// Elementwise fp32 -> bf16 conversion (prep pass; keeps GEMM loops cvt-free)
// ---------------------------------------------------------------------------
__global__ __launch_bounds__(256) void cvt_f32_bf16(const float* __restrict__ in,
                                                    __bf16* __restrict__ out, int n) {
  const int i = blockIdx.x * 256 + threadIdx.x;
  if (i < n) out[i] = (__bf16)in[i];
}

// ---------------------------------------------------------------------------
// Per-64-key-chunk mask summary: chunkok[b][c64] = 1 iff all 64 keys unmasked.
// Lets the flash hot loop skip all mask VALU with one uniform scalar load.
// ---------------------------------------------------------------------------
__global__ __launch_bounds__(128) void mask_chunks(const int* __restrict__ mask,
                                                   int* __restrict__ chunkok) {
  const int i = blockIdx.x * 128 + threadIdx.x;   // B * (LL/64) = 128 entries
  if (i >= BB * (LL / 64)) return;
  const int* p = mask + (i / (LL / 64)) * LL + (i % (LL / 64)) * 64;
  int ok = 1;
  for (int j = 0; j < 64; ++j) ok &= (p[j] != 0);
  chunkok[i] = ok;
}

// ---------------------------------------------------------------------------
// GEMM: out[M=8192,N=256] = A[8192,256] @ W[256,256]^T (+bias) (+addend)
// A and W are bf16 (pre-converted). Block = 4 waves; wave: 16(M) x 64(N).
// MODE 0: out = bf16(acc + bias + addend_f32)   (Ksum build)
// MODE 1: out = f32(acc + bias)                 (final projection)
// ---------------------------------------------------------------------------
template <int MODE>
__global__ __launch_bounds__(128) void gemm256(const __bf16* __restrict__ A,
                                               const __bf16* __restrict__ W,
                                               const float* __restrict__ bias,
                                               const float* __restrict__ addend,
                                               void* __restrict__ outp) {
  const int lane  = threadIdx.x & 31;
  const int wid   = threadIdx.x >> 5;
  const int mtile = blockIdx.x;
  const int lg    = lane & 15;
  const int hi    = lane >> 4;
  const int arow  = mtile * 16 + lg;
  const int kb    = hi ? 8 : 0;
  const int koff  = hi ? 16 : 0;
  const int n0    = wid * 64;

  v8f acc[4];
#pragma unroll
  for (int t = 0; t < 4; ++t) acc[t] = zero8();

  for (int kk = 0; kk < HH; kk += 32) {
    const v16bf afrag = af_from_bf16(A + (size_t)arow * HH + kk, kb);
#pragma unroll
    for (int t = 0; t < 4; ++t) {
      const int n = n0 + t * 16 + lg;
      const v16bf wfrag = bfrag_from_bf16(W + (size_t)n * HH + kk + koff);
      acc[t] = wmma_bf16(afrag, wfrag, acc[t]);
    }
  }

#pragma unroll
  for (int t = 0; t < 4; ++t) {
    const int n = n0 + t * 16 + lg;
    const float bv = bias[n];
#pragma unroll
    for (int r = 0; r < 8; ++r) {
      const int row = mtile * 16 + r + (hi ? 8 : 0);
      float val = acc[t][r] + bv;
      if constexpr (MODE == 0) {
        val += addend[(size_t)row * HH + n];
        ((__bf16*)outp)[(size_t)row * HH + n] = (__bf16)val;
      } else {
        ((float*)outp)[(size_t)row * HH + n] = val;
      }
    }
  }
}

// ---------------------------------------------------------------------------
// V transpose: v[B,L,H] fp32 -> vt[B*NH][DH][L] bf16 (keys contiguous per d)
// ---------------------------------------------------------------------------
__global__ __launch_bounds__(256) void transpose_v(const float* __restrict__ v,
                                                   __bf16* __restrict__ vt) {
  const int tid = blockIdx.x * 256 + threadIdx.x;
  const int d  = tid % DHH;
  int t1 = tid / DHH;
  const int h  = t1 % NHH;
  int t2 = t1 / NHH;
  const int l  = t2 % LL;
  const int b  = t2 / LL;
  vt[(((size_t)(b * NHH + h) * DHH + d) * LL) + l] = (__bf16)v[tid];
}

// ---------------------------------------------------------------------------
// Flash attention, transposed-score orientation, 64-key chunks, log2 domain.
//   scale_w/sqrt(DH)*log2e folded into the Q fragment (per-lane scalar)
//   S^T = Ksum (A) x Qscaled (B);  softmax via in-lane tree + one xor-16 shfl
//   O^T = Vt (A) x P^T (B);  P^T packed-exchanged in registers (no LDS)
// ---------------------------------------------------------------------------
__global__ __launch_bounds__(128) void flash_attn(const float* __restrict__ q,
                                                  const __bf16* __restrict__ ksum,
                                                  const __bf16* __restrict__ vt,
                                                  const int* __restrict__ mask,
                                                  const float* __restrict__ scale_w,
                                                  const int* __restrict__ chunkok,
                                                  __bf16* __restrict__ hidden) {
  const int lane  = threadIdx.x & 31;
  const int wid   = threadIdx.x >> 5;
  const int lg    = lane & 15;
  const int hi    = lane >> 4;
  const int hi8   = hi ? 8 : 0;
  const int kb    = hi ? 8 : 0;
  const int koff  = hi ? 16 : 0;
  const int qtile = blockIdx.x * 4 + wid;     // 128 q-tiles per (b,h)
  const int bh    = blockIdx.y;               // B*NH = 32
  const int b     = bh / NHH;
  const int h     = bh % NHH;
  const int query = qtile * 16 + lg;

  // fold 1/sqrt(DH) * scale_w[q] * log2(e) into Q (B-fragment, per-lane scale)
  const float rs = scale_w[h * LL + query] * (0.17677669529663688f * LOG2E);
  v16bf qf;
  {
    const float* qp = q + ((size_t)(b * LL + query) * HH + h * DHH + koff);
    v4f x0 = *(const v4f*)(qp);
    v4f x1 = *(const v4f*)(qp + 4);
    v4f x2 = *(const v4f*)(qp + 8);
    v4f x3 = *(const v4f*)(qp + 12);
#pragma unroll
    for (int i = 0; i < 4; ++i) {
      qf[i]      = (__bf16)(x0[i] * rs);
      qf[4 + i]  = (__bf16)(x1[i] * rs);
      qf[8 + i]  = (__bf16)(x2[i] * rs);
      qf[12 + i] = (__bf16)(x3[i] * rs);
    }
  }

  float mi = -1e30f, li = 0.f;
  v8f o0 = zero8(), o1 = zero8();             // O^T tiles: d 0-15 / 16-31

  const __bf16* kbb   = ksum + (size_t)b * LL * HH + h * DHH;
  const __bf16* vbase = vt + (size_t)bh * DHH * LL;
  const int*    mrow  = mask + b * LL;
  const int*    ckrow = chunkok + b * (LL / 64);

  for (int c = 0; c < LL; c += 64) {
    // 4 score tiles: keys c+t*16 .. c+t*16+15 (lane = query, VGPRs = keys)
    v8f s[4];
#pragma unroll
    for (int t = 0; t < 4; ++t) {
      const v16bf kf = af_from_bf16(kbb + (size_t)(c + t * 16 + lg) * HH, kb);
      s[t] = wmma_bf16(kf, qf, zero8());
    }

    float p[4][8];
#pragma unroll
    for (int t = 0; t < 4; ++t)
#pragma unroll
      for (int r = 0; r < 8; ++r) p[t][r] = s[t][r];

    // cold path only if this 64-key chunk contains masked keys (uniform branch)
    if (!ckrow[c >> 6]) {
#pragma unroll
      for (int t = 0; t < 4; ++t) {
        const int4 m0 = *(const int4*)(mrow + c + t * 16 + hi8);
        const int4 m1 = *(const int4*)(mrow + c + t * 16 + hi8 + 4);
        const int mm[8] = {m0.x, m0.y, m0.z, m0.w, m1.x, m1.y, m1.z, m1.w};
#pragma unroll
        for (int r = 0; r < 8; ++r) p[t][r] = mm[r] ? p[t][r] : MASKED_LOG2;
      }
    }

    // chunk max: in-lane tree + one cross-half shuffle
    float mx = p[0][0];
#pragma unroll
    for (int t = 0; t < 4; ++t)
#pragma unroll
      for (int r = (t ? 0 : 1); r < 8; ++r) mx = fmaxf(mx, p[t][r]);
    mx = fmaxf(mx, __shfl_xor(mx, 16, 32));

    const float mnew  = fmaxf(mi, mx);
    const float alpha = exp2f(mi - mnew);
    float sum = 0.f;
#pragma unroll
    for (int t = 0; t < 4; ++t)
#pragma unroll
      for (int r = 0; r < 8; ++r) {
        p[t][r] = exp2f(p[t][r] - mnew);
        sum += p[t][r];
      }
    sum += __shfl_xor(sum, 16, 32);
    li = li * alpha + sum;
    mi = mnew;
#pragma unroll
    for (int r = 0; r < 8; ++r) { o0[r] *= alpha; o1[r] *= alpha; }

    // P^T fragments: keys c..c+31 (tiles 0,1) and c+32..c+63 (tiles 2,3)
    const v16bf pf0 = make_pf(p[0], p[1], hi);
    const v16bf pf1 = make_pf(p[2], p[3], hi);

    // Vt rows as A-fragments; accumulate O^T over 64 keys (2 k-steps)
    const v16bf va0 = af_from_bf16(vbase + (size_t)lg * LL + c, kb);
    const v16bf va1 = af_from_bf16(vbase + (size_t)lg * LL + c + 32, kb);
    const v16bf vb0 = af_from_bf16(vbase + (size_t)(16 + lg) * LL + c, kb);
    const v16bf vb1 = af_from_bf16(vbase + (size_t)(16 + lg) * LL + c + 32, kb);
    o0 = wmma_bf16(va0, pf0, o0);
    o0 = wmma_bf16(va1, pf1, o0);
    o1 = wmma_bf16(vb0, pf0, o1);
    o1 = wmma_bf16(vb1, pf1, o1);
  }

  const float inv = 1.f / li;
  v8bf w0, w1;
#pragma unroll
  for (int r = 0; r < 8; ++r) {
    w0[r] = (__bf16)(o0[r] * inv);
    w1[r] = (__bf16)(o1[r] * inv);
  }
  __bf16* hrow = hidden + (size_t)(b * LL + query) * HH + h * DHH;
  *(v8bf*)(hrow + hi8)      = w0;
  *(v8bf*)(hrow + 16 + hi8) = w1;
}

// ---------------------------------------------------------------------------
extern "C" void kernel_launch(void* const* d_in, const int* in_sizes, int n_in,
                              void* d_out, int out_size, void* d_ws, size_t ws_size,
                              hipStream_t stream) {
  const float* q       = (const float*)d_in[0];
  const float* k       = (const float*)d_in[1];
  const float* v       = (const float*)d_in[2];
  const float* k_b     = (const float*)d_in[3];
  const int*   mask    = (const int*)d_in[4];
  const float* scale_w = (const float*)d_in[5];
  const float* Wb      = (const float*)d_in[6];
  const float* bb      = (const float*)d_in[7];
  const float* Ww      = (const float*)d_in[8];
  const float* bw      = (const float*)d_in[9];

  char* ws = (char*)d_ws;
  const size_t tb = (size_t)BB * LL * HH * sizeof(__bf16);   // 4 MB
  const size_t wb = (size_t)HH * HH * sizeof(__bf16);        // 128 KB
  __bf16* ksum   = (__bf16*)(ws);
  __bf16* vt     = (__bf16*)(ws + tb);
  __bf16* hidden = (__bf16*)(ws + 2 * tb);
  __bf16* kb_bf  = (__bf16*)(ws + 3 * tb);
  __bf16* wb_bf  = (__bf16*)(ws + 4 * tb);
  __bf16* ww_bf  = (__bf16*)(ws + 4 * tb + wb);
  int*    ckok   = (int*)(ws + 4 * tb + 2 * wb);

  const int nKB = BB * LL * HH;   // 2M
  const int nW  = HH * HH;        // 64K

  // prep
  cvt_f32_bf16<<<dim3((nKB + 255) / 256), dim3(256), 0, stream>>>(k_b, kb_bf, nKB);
  cvt_f32_bf16<<<dim3((nW + 255) / 256), dim3(256), 0, stream>>>(Wb, wb_bf, nW);
  cvt_f32_bf16<<<dim3((nW + 255) / 256), dim3(256), 0, stream>>>(Ww, ww_bf, nW);
  mask_chunks<<<dim3(1), dim3(128), 0, stream>>>(mask, ckok);

  // 1) Ksum = bf16(k + k_b @ Wb^T + bb)  [merges the two QK^T matmuls]
  gemm256<0><<<dim3(BB * LL / 16), dim3(128), 0, stream>>>(kb_bf, wb_bf, bb, k,
                                                           (void*)ksum);

  // 2) Vt[bh][d][l] for contiguous P@V A-fragments
  transpose_v<<<dim3(BB * LL * HH / 256), dim3(256), 0, stream>>>(v, vt);

  // 3) attention: bf16 WMMA, f32 accum, log2-domain online softmax
  flash_attn<<<dim3(LL / 16 / 4, BB * NHH), dim3(128), 0, stream>>>(
      q, ksum, vt, mask, scale_w, ckok, hidden);

  // 4) out = hidden @ Ww^T + bw (fp32)
  gemm256<1><<<dim3(BB * LL / 16), dim3(128), 0, stream>>>(hidden, ww_bf, bw,
                                                           nullptr, d_out);
}
